// FusedFP8Linear_37666863186535
// MI455X (gfx1250) — 8-bit matrix operand order repaired and checked by reading, compiled, not run on hardware
//
#include <hip/hip_runtime.h>
#include <hip/hip_bf16.h>
#include <cstdint>

typedef __attribute__((ext_vector_type(16))) int   v16i;
typedef __attribute__((ext_vector_type(4)))  int   v4i;
typedef __attribute__((ext_vector_type(8)))  float v8f;

#define AS1 __attribute__((address_space(1)))
#define AS3 __attribute__((address_space(3)))

static constexpr int MDIM = 32768;
static constexpr int NDIM = 1024;
static constexpr int KDIM = 1024;
static constexpr int HLEN = 16;
static constexpr float E4M3_MAX = 448.0f;

__global__ void prep_scales_kernel(const float* __restrict__ xh,
                                   const float* __restrict__ wh,
                                   float* __restrict__ sc,
                                   unsigned* __restrict__ am) {
    if (threadIdx.x == 0) {
        float mx = 0.f, mw = 0.f;
        for (int i = 0; i < HLEN; ++i) { mx = fmaxf(mx, xh[i]); mw = fmaxf(mw, wh[i]); }
        float sx = (mx > 0.f) ? (mx / E4M3_MAX) : 1.0f;
        float sw = (mw > 0.f) ? (mw / E4M3_MAX) : 1.0f;
        sc[0] = sx; sc[1] = sw; sc[2] = 1.0f / sx; sc[3] = 1.0f / sw;
        am[0] = 0u; am[1] = 0u;
    }
}

__global__ void quant_kernel(const __hip_bfloat16* __restrict__ in,
                             uint8_t* __restrict__ out,
                             const float* __restrict__ sc, int inv_idx,
                             unsigned* __restrict__ amax,
                             long long n8) {
    long long g = (long long)blockIdx.x * blockDim.x + threadIdx.x;
    float inv = sc[inv_idx];
    float lmax = 0.f;
    if (g < n8) {
        const uint4 v = ((const uint4*)in)[g];
        unsigned w[4] = {v.x, v.y, v.z, v.w};
        float f[8];
#pragma unroll
        for (int i = 0; i < 4; ++i) {
            unsigned lo = w[i] << 16;
            unsigned hi = w[i] & 0xffff0000u;
            f[2 * i + 0] = __builtin_bit_cast(float, lo);
            f[2 * i + 1] = __builtin_bit_cast(float, hi);
        }
#pragma unroll
        for (int i = 0; i < 8; ++i) lmax = fmaxf(lmax, fabsf(f[i]));
        unsigned q[2];
#pragma unroll
        for (int p = 0; p < 2; ++p) {
            float a = fminf(fmaxf(f[4 * p + 0] * inv, -E4M3_MAX), E4M3_MAX);
            float b = fminf(fmaxf(f[4 * p + 1] * inv, -E4M3_MAX), E4M3_MAX);
            float c = fminf(fmaxf(f[4 * p + 2] * inv, -E4M3_MAX), E4M3_MAX);
            float d = fminf(fmaxf(f[4 * p + 3] * inv, -E4M3_MAX), E4M3_MAX);
            int r = __builtin_amdgcn_cvt_pk_fp8_f32(a, b, 0, false);
            r = __builtin_amdgcn_cvt_pk_fp8_f32(c, d, r, true);
            q[p] = (unsigned)r;
        }
        ((uint2*)out)[g] = make_uint2(q[0], q[1]);
    }
#pragma unroll
    for (int off = 16; off > 0; off >>= 1)
        lmax = fmaxf(lmax, __shfl_xor(lmax, off, 32));
    if ((threadIdx.x & 31) == 0) atomicMax(amax, __float_as_uint(lmax));
}

static constexpr int PAD = 144;

__global__ void __launch_bounds__(256)
fp8_gemm_kernel(const uint8_t* __restrict__ x8, const uint8_t* __restrict__ w8,
                const __hip_bfloat16* __restrict__ bias,
                const float* __restrict__ sc,
                __hip_bfloat16* __restrict__ y) {
    __shared__ __align__(16) uint8_t smW[2][128 * PAD];
    __shared__ __align__(16) uint8_t smX[2][128 * PAD];

    const int tid   = threadIdx.x;
    const int wave  = tid >> 5;
    const int lane  = tid & 31;
    const int waveM = wave >> 1;
    const int waveN = wave & 1;
    const int half  = lane >> 4;
    const int l16   = lane & 15;

    const int tileM = blockIdx.y * 128;
    const int tileN = blockIdx.x * 128;

    const uint8_t* gW = w8 + (size_t)tileN * KDIM;
    const uint8_t* gX = x8 + (size_t)tileM * KDIM;

    auto stage = [&](int buf, int ksByte) {
#pragma unroll
        for (int i = 0; i < 4; ++i) {
            int chunk = tid + 256 * i;
            int r = chunk >> 3;
            int c = (chunk & 7) * 16;
            __builtin_amdgcn_global_load_async_to_lds_b128(
                (AS1 v4i*)(gW + (size_t)r * KDIM + ksByte + c),
                (AS3 v4i*)(&smW[buf][r * PAD + c]), 0, 0);
            __builtin_amdgcn_global_load_async_to_lds_b128(
                (AS1 v4i*)(gX + (size_t)r * KDIM + ksByte + c),
                (AS3 v4i*)(&smX[buf][r * PAD + c]), 0, 0);
        }
    };

    v8f acc[2][4];
#pragma unroll
    for (int mi = 0; mi < 2; ++mi)
#pragma unroll
        for (int ni = 0; ni < 4; ++ni)
#pragma unroll
            for (int r = 0; r < 8; ++r) acc[mi][ni][r] = 0.f;

    const float outscale = sc[0] * sc[1];

    stage(0, 0);
    for (int ks = 0; ks < KDIM / 128; ++ks) {
        const int cur = ks & 1;
        if (ks + 1 < KDIM / 128) {
            stage(cur ^ 1, (ks + 1) * 128);
            asm volatile("s_wait_asynccnt 0x8" ::: "memory");
        } else {
            asm volatile("s_wait_asynccnt 0x0" ::: "memory");
        }
        __syncthreads();

        v16i af[4];
#pragma unroll
        for (int ni = 0; ni < 4; ++ni) {
            const uint8_t* ap = &smW[cur][(waveN * 64 + ni * 16 + l16) * PAD];
#pragma unroll
            for (int s = 0; s < 2; ++s)
#pragma unroll
                for (int c = 0; c < 4; ++c) {
                    uint2 t = *(const uint2*)(ap + s * 64 + c * 16 + half * 8);
                    af[ni][8 * s + 2 * c + 0] = (int)t.x;
                    af[ni][8 * s + 2 * c + 1] = (int)t.y;
                }
        }
        v16i bf[2];
#pragma unroll
        for (int mi = 0; mi < 2; ++mi) {
            const uint8_t* bp = &smX[cur][(waveM * 32 + mi * 16 + l16) * PAD];
#pragma unroll
            for (int c = 0; c < 4; ++c) {
                uint2 t0 = *(const uint2*)(bp + c * 32 + half * 8);
                uint2 t1 = *(const uint2*)(bp + c * 32 + 16 + half * 8);
                bf[mi][4 * c + 0] = (int)t0.x;
                bf[mi][4 * c + 1] = (int)t0.y;
                bf[mi][4 * c + 2] = (int)t1.x;
                bf[mi][4 * c + 3] = (int)t1.y;
            }
        }
#pragma unroll
        for (int mi = 0; mi < 2; ++mi)
#pragma unroll
            for (int ni = 0; ni < 4; ++ni)
                acc[mi][ni] = __builtin_amdgcn_wmma_f32_16x16x128_fp8_fp8(
                    af[ni], bf[mi], (short)0, acc[mi][ni], false, false);

        __syncthreads();
    }

#pragma unroll
    for (int ni = 0; ni < 4; ++ni) {
        const int n0 = tileN + waveN * 64 + ni * 16 + 8 * half;
        const uint4 bb = *(const uint4*)(bias + n0);
        unsigned bw[4] = {bb.x, bb.y, bb.z, bb.w};
        float bfv[8];
#pragma unroll
        for (int i = 0; i < 4; ++i) {
            unsigned lo = bw[i] << 16, hi = bw[i] & 0xffff0000u;
            bfv[2 * i + 0] = __builtin_bit_cast(float, lo);
            bfv[2 * i + 1] = __builtin_bit_cast(float, hi);
        }
#pragma unroll
        for (int mi = 0; mi < 2; ++mi) {
            const int m = tileM + waveM * 32 + mi * 16 + l16;
            unsigned ow[4];
#pragma unroll
            for (int p = 0; p < 4; ++p) {
                float v0 = acc[mi][ni][2 * p + 0] * outscale + bfv[2 * p + 0];
                float v1 = acc[mi][ni][2 * p + 1] * outscale + bfv[2 * p + 1];
                unsigned short b0 = __builtin_bit_cast(unsigned short, __float2bfloat16(v0));
                unsigned short b1 = __builtin_bit_cast(unsigned short, __float2bfloat16(v1));
                ow[p] = (unsigned)b0 | ((unsigned)b1 << 16);
            }
            *(uint4*)(y + (size_t)m * NDIM + n0) = make_uint4(ow[0], ow[1], ow[2], ow[3]);
        }
    }
}

__global__ void finalize_hist_kernel(const float* __restrict__ xh,
                                     const float* __restrict__ wh,
                                     const unsigned* __restrict__ am,
                                     float* __restrict__ outxh,
                                     float* __restrict__ outwh) {
    int i = threadIdx.x;
    if (i < HLEN) {
        outxh[i] = (i < HLEN - 1) ? xh[i + 1] : __uint_as_float(am[0]);
        outwh[i] = (i < HLEN - 1) ? wh[i + 1] : __uint_as_float(am[1]);
    }
}

extern "C" void kernel_launch(void* const* d_in, const int* in_sizes, int n_in,
                              void* d_out, int out_size, void* d_ws, size_t ws_size,
                              hipStream_t stream) {
    const __hip_bfloat16* x    = (const __hip_bfloat16*)d_in[0];
    const __hip_bfloat16* w    = (const __hip_bfloat16*)d_in[1];
    const __hip_bfloat16* bias = (const __hip_bfloat16*)d_in[2];
    const float* xh = (const float*)d_in[3];
    const float* wh = (const float*)d_in[4];

    uint8_t* ws = (uint8_t*)d_ws;
    uint8_t* x8 = ws;
    uint8_t* w8 = ws + (size_t)MDIM * KDIM;
    float*   sc = (float*)(ws + (size_t)MDIM * KDIM + (size_t)NDIM * KDIM);
    unsigned* am = (unsigned*)(sc + 4);

    __hip_bfloat16* y = (__hip_bfloat16*)d_out;
    float* outxh = (float*)((char*)d_out + (size_t)MDIM * NDIM * sizeof(__hip_bfloat16));
    float* outwh = outxh + HLEN;

    prep_scales_kernel<<<1, 32, 0, stream>>>(xh, wh, sc, am);

    long long nx8 = (long long)MDIM * KDIM / 8;
    long long nw8 = (long long)NDIM * KDIM / 8;
    quant_kernel<<<dim3((unsigned)(nx8 / 256)), 256, 0, stream>>>(x, x8, sc, 2, am + 0, nx8);
    quant_kernel<<<dim3((unsigned)(nw8 / 256)), 256, 0, stream>>>(w, w8, sc, 3, am + 1, nw8);

    fp8_gemm_kernel<<<dim3(NDIM / 128, MDIM / 128), 256, 0, stream>>>(x8, w8, bias, sc, y);

    finalize_hist_kernel<<<1, 32, 0, stream>>>(xh, wh, am, outxh, outwh);
}
